// Structure_Aggregation_61340722921490
// MI455X (gfx1250) — compile-verified, matrix operand order checked by reading, not run on hardware
//
#include <hip/hip_runtime.h>
#include <hip/hip_bf16.h>

// ---------------------------------------------------------------------------
// Structure-aggregation network for MI455X (gfx1250, wave32, WMMA).
// All conv3x3 blocks run as implicit GEMM on v_wmma_f32_16x16x32_f16:
//   M = 64 output pixels per workgroup (4 WMMA sub-tiles, 4 accumulators/wave)
//   N = 16 output channels per wave (Npad/16 waves per workgroup)
//   K = ci*9 (padded to a multiple of 32)
// Activations are f16 planar [B][C][H][W]; accumulation f32.
// GELU + LayerNorm (channel) fused in LDS; the last layer additionally fuses
// softmax-over-128-disparities + rolled-guidance aggregation.
// ---------------------------------------------------------------------------

typedef __attribute__((ext_vector_type(16))) _Float16 v16h;
typedef __attribute__((ext_vector_type(8)))  float    v8f;

#define N_LAYERS 17
#define MTILES 4                 // 4 x 16-pixel M sub-tiles per workgroup
#define MPIX  (MTILES * 16)      // 64 pixels per workgroup

__device__ __forceinline__ float gelu_exact(float x) {
    return 0.5f * x * (1.0f + erff(x * 0.70710678118654752f));
}

// ---- weight packing: f32 [co][ci][3][3] -> f16 WMMA B-fragment order ------
// dst flat index = ((nt*(Kpad/32) + q)*32 + L)*16 + j   maps to B[k][n] with
//   n = nt*16 + (L & 15)
//   k = q*32 + (L>>4)*8 + (j < 8 ? j : j + 8)
// (mirror of the ISA 16-bit 16x32 A-matrix lane layout)
__global__ void pack_weights_kernel(const float* __restrict__ w,
                                    _Float16* __restrict__ dst,
                                    int ci, int co, int Kpad, int Npad) {
    int idx = blockIdx.x * blockDim.x + threadIdx.x;
    int total = Npad * Kpad;
    if (idx >= total) return;
    int perNT = Kpad * 16;
    int nt  = idx / perNT;
    int rem = idx - nt * perNT;
    int q = rem >> 9;          // /512 halves per 32-deep K chunk
    int r = rem & 511;
    int L = r >> 4;
    int j = r & 15;
    int n = nt * 16 + (L & 15);
    int k = q * 32 + ((L >> 4) << 3) + (j < 8 ? j : j + 8);
    float v = 0.0f;
    int K = ci * 9;
    if (k < K && n < co) {
        int c = k / 9, tap = k - c * 9;
        v = w[(n * ci + c) * 9 + tap];
    }
    dst[idx] = (_Float16)v;
}

// ---- 2x2 max-pool ---------------------------------------------------------
__global__ void maxpool2_f32_to_f16(const float* __restrict__ in,
                                    _Float16* __restrict__ out,
                                    int B, int C, int H, int W) {
    int ho = H >> 1, wo = W >> 1;
    long total = (long)B * C * ho * wo;
    long idx = (long)blockIdx.x * blockDim.x + threadIdx.x;
    if (idx >= total) return;
    int x = (int)(idx % wo); long t = idx / wo;
    int y = (int)(t % ho);   t /= ho;
    int c = (int)(t % C);    int b = (int)(t / C);
    const float* p = in + ((((long)b * C + c) * H + 2 * y) * W + 2 * x);
    float m = fmaxf(fmaxf(p[0], p[1]), fmaxf(p[W], p[W + 1]));
    out[idx] = (_Float16)m;
}

__global__ void maxpool2_f16(const _Float16* __restrict__ in,
                             _Float16* __restrict__ out,
                             int B, int C, int H, int W) {
    int ho = H >> 1, wo = W >> 1;
    long total = (long)B * C * ho * wo;
    long idx = (long)blockIdx.x * blockDim.x + threadIdx.x;
    if (idx >= total) return;
    int x = (int)(idx % wo); long t = idx / wo;
    int y = (int)(t % ho);   t /= ho;
    int c = (int)(t % C);    int b = (int)(t / C);
    const _Float16* p = in + ((((long)b * C + c) * H + 2 * y) * W + 2 * x);
    float m = fmaxf(fmaxf((float)p[0], (float)p[1]),
                    fmaxf((float)p[W], (float)p[W + 1]));
    out[idx] = (_Float16)m;
}

// ---- bilinear x2 upsample, align_corners=True -----------------------------
__global__ void upsample2x_f16(const _Float16* __restrict__ in,
                               _Float16* __restrict__ out,
                               int B, int C, int h, int w) {
    int ho = 2 * h, wo = 2 * w;
    long total = (long)B * C * ho * wo;
    long idx = (long)blockIdx.x * blockDim.x + threadIdx.x;
    if (idx >= total) return;
    int x = (int)(idx % wo); long t = idx / wo;
    int y = (int)(t % ho);   t /= ho;
    int c = (int)(t % C);    int b = (int)(t / C);
    float sy = (float)(h - 1) / (float)(ho - 1);
    float sx = (float)(w - 1) / (float)(wo - 1);
    float py = y * sy, px = x * sx;
    int y0 = (int)py, x0 = (int)px;
    int y1 = min(y0 + 1, h - 1), x1 = min(x0 + 1, w - 1);
    float ty = py - (float)y0, tx = px - (float)x0;
    const _Float16* p = in + (((long)b * C + c) * h) * (long)w;
    float v00 = (float)p[(long)y0 * w + x0];
    float v01 = (float)p[(long)y0 * w + x1];
    float v10 = (float)p[(long)y1 * w + x0];
    float v11 = (float)p[(long)y1 * w + x1];
    float v0 = v00 + (v01 - v00) * tx;
    float v1 = v10 + (v11 - v10) * tx;
    out[idx] = (_Float16)(v0 + (v1 - v0) * ty);
}

// ---- per-pixel correlation over 24 channels, 32 disparities ---------------
__global__ void corr_kernel(const _Float16* __restrict__ ft,
                            const _Float16* __restrict__ fg,
                            _Float16* __restrict__ out,
                            int B, int C, int h, int w, int D) {
    long total = (long)B * D * h * w;
    long idx = (long)blockIdx.x * blockDim.x + threadIdx.x;
    if (idx >= total) return;
    int x = (int)(idx % w); long t = idx / w;
    int y = (int)(t % h);   t /= h;
    int d = (int)(t % D);   int b = (int)(t / D);
    int xs = x - d; if (xs < 0) xs += w;
    long plane = (long)h * w;
    long base = (long)b * C * plane + (long)y * w;
    float acc = 0.0f;
    for (int c = 0; c < C; ++c)
        acc += (float)ft[base + c * plane + x] * (float)fg[base + c * plane + xs];
    out[idx] = (_Float16)acc;
}

// ---- conv3x3 (implicit GEMM, WMMA) + GELU + LayerNorm [+ softmax/agg] -----
// grid: (H*W/64, B).  block: (Npad/16)*32 threads, one N-tile per wave.
// Each wave runs 4 WMMAs per 32-deep K chunk (one per M sub-tile), reusing a
// single B fragment -> 4x lower weight-fragment traffic, dense XDL issue.
// mode 0: write f16 activations.  mode 1 (final layer, co==128): fused
// softmax over disparity channels + weighted sum with rolled guidance -> f32.
__global__ void conv3x3_gelu_ln_wmma(const _Float16* __restrict__ in,
                                     const _Float16* __restrict__ wpack,
                                     const float* __restrict__ gamma,
                                     const float* __restrict__ beta,
                                     _Float16* __restrict__ out,
                                     const float* __restrict__ guid,
                                     float* __restrict__ fout,
                                     int H, int W, int ci, int co,
                                     int Kpad, int Npad, int mode) {
    extern __shared__ char smem[];
    _Float16* ldsA = (_Float16*)smem;                 // [MTILES][16*Kpad], fragment order
    float* ldsC  = (float*)(smem + (size_t)MPIX * Kpad * 2);   // [MPIX*Npad]
    float* ldsMu = ldsC + MPIX * Npad;                          // [MPIX]
    float* ldsRs = ldsMu + MPIX;                                // [MPIX]

    const int tid = threadIdx.x;
    const int nthreads = blockDim.x;
    const int lane = tid & 31;
    const int wv = tid >> 5;                 // N-tile handled by this wave
    const int b = blockIdx.y;
    const int tileBase = blockIdx.x * MPIX;  // 64 pixels per workgroup
    const int K = ci * 9;
    const int nq = Kpad >> 5;                // number of 32-deep K chunks
    const int aStride = 16 * Kpad;           // halves per M sub-tile

    // Stage the 64-pixel input patch tile into LDS in A-fragment order, so a
    // lane's fragment is one contiguous 32B read (ds_load_b128 x2).
    for (int idx = tid; idx < MPIX * Kpad; idx += nthreads) {
        int s = idx / aStride;               // M sub-tile
        int rem = idx - s * aStride;
        int q = rem >> 9;
        int r = rem & 511;
        int L = r >> 4;
        int j = r & 15;
        int m = s * 16 + (L & 15);           // pixel within the 64
        int k = q * 32 + ((L >> 4) << 3) + (j < 8 ? j : j + 8);
        _Float16 v = (_Float16)0.0f;
        if (k < K) {
            int gp = tileBase + m;
            int y = gp / W, x = gp - y * W;
            int c = k / 9, tap = k - c * 9;
            int yy = y + tap / 3 - 1;
            int xx = x + (tap % 3) - 1;
            if (yy >= 0 && yy < H && xx >= 0 && xx < W)
                v = in[(((long)b * ci + c) * H + yy) * (long)W + xx];
        }
        ldsA[idx] = v;
    }
    __syncthreads();

    // Implicit GEMM main loop: 4x D(16x16,f32) += A_s(16x32,f16) x B(32x16,f16).
    v8f acc0 = {0.f,0.f,0.f,0.f,0.f,0.f,0.f,0.f};
    v8f acc1 = acc0, acc2 = acc0, acc3 = acc0;
    const _Float16* wp = wpack + ((size_t)wv * nq) * 512;
    for (int q = 0; q < nq; ++q) {
        const _Float16* bp = wp + ((q * 32 + lane) << 4);
        v16h bm = *(const v16h*)bp;
        __builtin_prefetch(bp + 512, 0, 1);   // next K chunk (speculative-safe)
        const _Float16* ap = ldsA + ((q * 32 + lane) << 4);
        v16h a0 = *(const v16h*)(ap);
        v16h a1 = *(const v16h*)(ap + aStride);
        v16h a2 = *(const v16h*)(ap + 2 * aStride);
        v16h a3 = *(const v16h*)(ap + 3 * aStride);
        acc0 = __builtin_amdgcn_wmma_f32_16x16x32_f16(false, a0, false, bm,
                                                      (short)0, acc0, false, false);
        acc1 = __builtin_amdgcn_wmma_f32_16x16x32_f16(false, a1, false, bm,
                                                      (short)0, acc1, false, false);
        acc2 = __builtin_amdgcn_wmma_f32_16x16x32_f16(false, a2, false, bm,
                                                      (short)0, acc2, false, false);
        acc3 = __builtin_amdgcn_wmma_f32_16x16x32_f16(false, a3, false, bm,
                                                      (short)0, acc3, false, false);
    }

    // GELU, stage results for LayerNorm. C/D layout: VGPR r holds
    // M = r + 8*(lane>>4), N = lane&15.
    {
        int nl = (wv << 4) + (lane & 15);
        int mh = (lane >> 4) << 3;
#pragma unroll
        for (int r = 0; r < 8; ++r) {
            ldsC[(0 * 16 + mh + r) * Npad + nl] = gelu_exact(acc0[r]);
            ldsC[(1 * 16 + mh + r) * Npad + nl] = gelu_exact(acc1[r]);
            ldsC[(2 * 16 + mh + r) * Npad + nl] = gelu_exact(acc2[r]);
            ldsC[(3 * 16 + mh + r) * Npad + nl] = gelu_exact(acc3[r]);
        }
    }
    __syncthreads();

    // Per-pixel LayerNorm statistics over the co real channels (biased var).
    if (tid < MPIX) {
        float s = 0.f, s2 = 0.f;
        for (int n = 0; n < co; ++n) {
            float v = ldsC[tid * Npad + n];
            s += v; s2 += v * v;
        }
        float mu = s / (float)co;
        float var = s2 / (float)co - mu * mu;
        ldsMu[tid] = mu;
        ldsRs[tid] = rsqrtf(var + 1e-6f);
    }
    __syncthreads();

    if (mode == 0) {
        for (int idx = tid; idx < MPIX * co; idx += nthreads) {
            int m = idx / co, n = idx - m * co;
            float v = gamma[n] * ((ldsC[m * Npad + n] - ldsMu[m]) * ldsRs[m]) + beta[n];
            int gp = tileBase + m;
            int y = gp / W, x = gp - y * W;
            out[(((long)b * co + n) * H + y) * (long)W + x] = (_Float16)v;
        }
    } else {
        // Fused: LN -> softmax over co==128 disparity channels -> weighted sum
        // with guidance rolled right by d along width.
        if (tid < MPIX) {
            int m = tid;
            int gp = tileBase + m;
            int y = gp / W, x = gp - y * W;
            float mx = -1e30f;
            for (int n = 0; n < co; ++n) {
                float v = gamma[n] * ((ldsC[m * Npad + n] - ldsMu[m]) * ldsRs[m]) + beta[n];
                ldsC[m * Npad + n] = v;
                mx = fmaxf(mx, v);
            }
            const float* grow = guid + ((long)b * H + y) * (long)W;
            float se = 0.f, sw = 0.f;
            for (int d = 0; d < co; ++d) {
                float e = __expf(ldsC[m * Npad + d] - mx);
                int xs = x - d; if (xs < 0) xs += W;
                se += e;
                sw += e * grow[xs];
            }
            fout[((long)b * H + y) * (long)W + x] = sw / se;
        }
    }
}

// ---------------------------------------------------------------------------
extern "C" void kernel_launch(void* const* d_in, const int* in_sizes, int n_in,
                              void* d_out, int out_size, void* d_ws, size_t ws_size,
                              hipStream_t stream) {
    (void)in_sizes; (void)n_in; (void)out_size; (void)ws_size;

    const float* noisy = (const float*)d_in[0];
    const float* guid  = (const float*)d_in[1];
    // params flattened in dict order: t1..t6, g1..g6, w1..w5; each {w,g,b}
    auto Wp = [&](int l) { return (const float*)d_in[2 + 3 * l + 0]; };
    auto Gp = [&](int l) { return (const float*)d_in[2 + 3 * l + 1]; };
    auto Bp = [&](int l) { return (const float*)d_in[2 + 3 * l + 2]; };

    struct LC { int ci, co; };
    const LC lc[N_LAYERS] = {
        {1,48},{48,96},{96,96},{96,96},{96,96},{96,24},     // t1..t6
        {1,48},{48,96},{96,96},{96,96},{96,96},{96,24},     // g1..g6
        {32,96},{96,96},{96,96},{96,96},{96,128}            // w1..w5
    };
    auto rup = [](int v, int m) { return (v + m - 1) / m * m; };

    // Workspace layout (all f16 halves).
    _Float16* ws = (_Float16*)d_ws;
    const size_t WSLOT = (size_t)128 * 864;            // max packed layer (w5)
    const size_t BIG   = (size_t)2 * 96 * 256 * 512;   // biggest activation
    _Float16* wpck = ws;
    _Float16* bufA = ws + N_LAYERS * WSLOT;
    _Float16* bufB = bufA + BIG;
    _Float16* FT   = bufB + BIG;                       // [2,24,64,128]
    _Float16* FG   = FT + (size_t)2 * 24 * 64 * 128;
    _Float16* CR   = FG + (size_t)2 * 24 * 64 * 128;   // [2,32,64,128]

    // Pack all conv weights into WMMA B-fragment order (deterministic, every call).
    for (int l = 0; l < N_LAYERS; ++l) {
        int Kpad = rup(lc[l].ci * 9, 32), Npad = rup(lc[l].co, 16);
        int total = Kpad * Npad;
        pack_weights_kernel<<<(total + 255) / 256, 256, 0, stream>>>(
            Wp(l), wpck + l * WSLOT, lc[l].ci, lc[l].co, Kpad, Npad);
    }

    auto conv = [&](int l, const _Float16* in, _Float16* out, int H, int W,
                    int mode, float* fo) {
        int ci = lc[l].ci, co = lc[l].co;
        int Kpad = rup(ci * 9, 32), Npad = rup(co, 16);
        int NT = Npad / 16;
        dim3 grid((unsigned)(H * W / MPIX), 2);
        size_t lds = (size_t)MPIX * Kpad * 2 + (size_t)MPIX * Npad * 4
                   + 2 * MPIX * 4;
        conv3x3_gelu_ln_wmma<<<grid, NT * 32, lds, stream>>>(
            in, wpck + l * WSLOT, Gp(l), Bp(l), out, guid, fo,
            H, W, ci, co, Kpad, Npad, mode);
    };

    long t;
    // ---- target (noisy) branch -> FT [2,24,64,128] ----
    t = 2L * 1 * 128 * 256;
    maxpool2_f32_to_f16<<<(unsigned)((t + 255) / 256), 256, 0, stream>>>(
        noisy, bufA, 2, 1, 256, 512);
    conv(0, bufA, bufB, 128, 256, 0, nullptr);           // t1: 1->48
    t = 2L * 48 * 64 * 128;
    maxpool2_f16<<<(unsigned)((t + 255) / 256), 256, 0, stream>>>(
        bufB, bufA, 2, 48, 128, 256);
    conv(1, bufA, bufB, 64, 128, 0, nullptr);            // t2: 48->96
    conv(2, bufB, bufA, 64, 128, 0, nullptr);            // t3
    conv(3, bufA, bufB, 64, 128, 0, nullptr);            // t4
    conv(4, bufB, bufA, 64, 128, 0, nullptr);            // t5
    conv(5, bufA, FT,   64, 128, 0, nullptr);            // t6: 96->24

    // ---- guidance branch -> FG [2,24,64,128] ----
    t = 2L * 1 * 128 * 256;
    maxpool2_f32_to_f16<<<(unsigned)((t + 255) / 256), 256, 0, stream>>>(
        guid, bufA, 2, 1, 256, 512);
    conv(6, bufA, bufB, 128, 256, 0, nullptr);           // g1
    t = 2L * 48 * 64 * 128;
    maxpool2_f16<<<(unsigned)((t + 255) / 256), 256, 0, stream>>>(
        bufB, bufA, 2, 48, 128, 256);
    conv(7,  bufA, bufB, 64, 128, 0, nullptr);           // g2
    conv(8,  bufB, bufA, 64, 128, 0, nullptr);           // g3
    conv(9,  bufA, bufB, 64, 128, 0, nullptr);           // g4
    conv(10, bufB, bufA, 64, 128, 0, nullptr);           // g5
    conv(11, bufA, FG,   64, 128, 0, nullptr);           // g6

    // ---- correlation volume [2,32,64,128] ----
    t = 2L * 32 * 64 * 128;
    corr_kernel<<<(unsigned)((t + 255) / 256), 256, 0, stream>>>(
        FT, FG, CR, 2, 24, 64, 128, 32);

    // ---- weight head ----
    t = 2L * 32 * 128 * 256;
    upsample2x_f16<<<(unsigned)((t + 255) / 256), 256, 0, stream>>>(
        CR, bufA, 2, 32, 64, 128);
    conv(12, bufA, bufB, 128, 256, 0, nullptr);          // w1: 32->96
    conv(13, bufB, bufA, 128, 256, 0, nullptr);          // w2
    conv(14, bufA, bufB, 128, 256, 0, nullptr);          // w3
    t = 2L * 96 * 256 * 512;
    upsample2x_f16<<<(unsigned)((t + 255) / 256), 256, 0, stream>>>(
        bufB, bufA, 2, 96, 128, 256);
    conv(15, bufA, bufB, 256, 512, 0, nullptr);          // w4
    // w5: 96->128, fused LN + softmax(128) + rolled-guidance aggregation.
    conv(16, bufB, nullptr, 256, 512, 1, (float*)d_out);
}
